// MPNN_wo_GRU_58926951301830
// MI455X (gfx1250) — matrix operand order repaired and checked by reading, not compiled, not run on hardware
//
#include <hip/hip_runtime.h>
#include <hip/hip_bf16.h>

typedef _Float16 v8h  __attribute__((ext_vector_type(8)));
typedef _Float16 v16h __attribute__((ext_vector_type(16)));
typedef float    v8f  __attribute__((ext_vector_type(8)));

#define NODE_IN 74
#define EDGE_IN 12
#define OUT 64
#define EHID 128
#define OUTSQ 4096
#define NUM_STEPS 6

// ---------------------------------------------------------------------------
// h[v,o] = relu(sum_i nf[v,i]*Wp[i,o] + bp[o])      (fp32, small)
// ---------------------------------------------------------------------------
__global__ void proj_nodes(const float* __restrict__ nf,
                           const float* __restrict__ Wp,
                           const float* __restrict__ bp,
                           float* __restrict__ h, int V) {
    int idx = blockIdx.x * blockDim.x + threadIdx.x;
    int v = idx >> 6, o = idx & 63;
    if (v >= V) return;
    float acc = bp[o];
    const float* row = nf + (size_t)v * NODE_IN;
    #pragma unroll
    for (int i = 0; i < NODE_IN; ++i)
        acc = fmaf(row[i], Wp[i * OUT + o], acc);
    h[(size_t)v * OUT + o] = fmaxf(acc, 0.0f);
}

// ---------------------------------------------------------------------------
// h1[e,j] = relu(sum_i ef[e,i]*We1[i,j] + be1[j])   -> f16 (WMMA A operand)
// ---------------------------------------------------------------------------
__global__ void edge_layer1(const float* __restrict__ ef,
                            const float* __restrict__ We1,
                            const float* __restrict__ be1,
                            _Float16* __restrict__ h1, int E) {
    int idx = blockIdx.x * blockDim.x + threadIdx.x;
    int e = idx >> 7, j = idx & 127;
    if (e >= E) return;
    float acc = be1[j];
    const float* row = ef + (size_t)e * EDGE_IN;
    #pragma unroll
    for (int i = 0; i < EDGE_IN; ++i)
        acc = fmaf(row[i], We1[i * EHID + j], acc);
    h1[(size_t)e * EHID + j] = (_Float16)fmaxf(acc, 0.0f);
}

// ---------------------------------------------------------------------------
// Bt[n,k] = (f16) We2[k,n]   (transpose so B fragments load contiguously)
// ---------------------------------------------------------------------------
__global__ void transpose_We2(const float* __restrict__ We2,
                              _Float16* __restrict__ Bt) {
    int idx = blockIdx.x * blockDim.x + threadIdx.x;   // 4096*128 threads
    int n = idx >> 7, k = idx & 127;
    Bt[idx] = (_Float16)We2[(size_t)k * OUTSQ + n];
}

// ---------------------------------------------------------------------------
// ew[E,4096] = h1[E,128] @ We2[128,4096] + be2   (WMMA f16 -> f32 -> f16)
// grid = (ceil(E/16), 8), block = 256 (8 waves). Wave computes a 16x64 strip.
// ---------------------------------------------------------------------------
__global__ void ew_gemm(const _Float16* __restrict__ A,   // [E,128]
                        const _Float16* __restrict__ Bt,  // [4096,128]
                        const float* __restrict__ be2,    // [4096]
                        _Float16* __restrict__ C,         // [E,4096]
                        int E) {
    __shared__ __align__(16) _Float16 As[16 * 128];       // 4 KB A tile
    const int tid = threadIdx.x;
    const int e0 = blockIdx.x * 16;

    // cooperative A-tile load: 16 rows x 128 halves = 4 KB contiguous
    ((v8h*)As)[tid] = ((const v8h*)(A + (size_t)e0 * 128))[tid];
    __syncthreads();

    const int wave = tid >> 5, lane = tid & 31;
    const int nl = lane & 15, g = lane >> 4;
    const int n0 = blockIdx.y * 512 + wave * 64;

    v8f acc[4] = {v8f{}, v8f{}, v8f{}, v8f{}};

    #pragma unroll
    for (int kb = 0; kb < 128; kb += 32) {
        // A fragment (16-bit A layout: VGPR v holds K = 2v + g*8, +16 for v>=4)
        const _Float16* ab = As + nl * 128 + kb + g * 8;
        v8h a0 = *(const v8h*)ab;            // K offsets g*8 .. g*8+7
        v8h a1 = *(const v8h*)(ab + 16);     // K offsets 16+g*8 .. 16+g*8+7
        v16h a = __builtin_shufflevector(a0, a1,
                 0,1,2,3,4,5,6,7,8,9,10,11,12,13,14,15);
        #pragma unroll
        for (int t = 0; t < 4; ++t) {
            // B fragment (same K order as A: K = 2v + g*8, +16 for v>=4)
            const _Float16* bb = Bt + (size_t)(n0 + t * 16 + nl) * 128 + kb + g * 8;
            v8h b0 = *(const v8h*)bb;
            v8h b1 = *(const v8h*)(bb + 16);
            v16h b = __builtin_shufflevector(b0, b1,
                     0,1,2,3,4,5,6,7,8,9,10,11,12,13,14,15);
            acc[t] = __builtin_amdgcn_wmma_f32_16x16x32_f16(
                         false, a, false, b, (short)0, acc[t], false, false);
        }
    }

    // epilogue: C/D layout -> lane holds (M = i + 8*g, N = nl)
    #pragma unroll
    for (int t = 0; t < 4; ++t) {
        int n = n0 + t * 16 + nl;
        float bias = be2[n];
        #pragma unroll
        for (int i = 0; i < 8; ++i) {
            int er = e0 + i + 8 * g;
            if (er < E)
                C[(size_t)er * OUTSQ + n] = (_Float16)(acc[t][i] + bias);
        }
    }
}

// ---------------------------------------------------------------------------
// zero fill
// ---------------------------------------------------------------------------
__global__ void zero_f32(float* __restrict__ p, int n) {
    int idx = blockIdx.x * blockDim.x + threadIdx.x;
    if (idx < n) p[idx] = 0.0f;
}

// ---------------------------------------------------------------------------
// Message + scatter: one wave per edge.
//   msg[e,o] = sum_i h[src[e],i] * ew[e, i*64+o];  atomicAdd into agg[dst[e]]
// Streams ew with b128 loads (4 rows / 512 B per instruction).
// ---------------------------------------------------------------------------
__global__ void msg_scatter(const float* __restrict__ h,
                            const _Float16* __restrict__ ew,
                            const int* __restrict__ src,
                            const int* __restrict__ dst,
                            float* __restrict__ agg, int E) {
    const int lane = threadIdx.x & 31;
    const int wave = (blockIdx.x * blockDim.x + threadIdx.x) >> 5;
    if (wave >= E) return;
    const int e = wave;

    const int s = src[e];
    // lane l holds h_src[l] and h_src[l+32]
    const float h0 = h[(size_t)s * OUT + lane];
    const float h1v = h[(size_t)s * OUT + 32 + lane];

    const _Float16* M = ew + (size_t)e * OUTSQ;
    const int sub = lane >> 3;          // which of 4 rows this lane covers
    const int c8  = (lane & 7) * 8;     // 8 output columns owned by this lane

    float acc[8];
    #pragma unroll
    for (int j = 0; j < 8; ++j) acc[j] = 0.0f;

    #pragma unroll
    for (int t = 0; t < 16; ++t) {
        const int row = 4 * t + sub;                       // 0..63
        // broadcast h_src[row] (lane-varying source -> ds_bpermute)
        const float hr = __shfl((t < 8) ? h0 : h1v, row & 31, 32);
        const v8h m = *(const v8h*)(M + (size_t)row * OUT + c8);  // b128
        #pragma unroll
        for (int j = 0; j < 8; ++j)
            acc[j] = fmaf(hr, (float)m[j], acc[j]);
    }

    // reduce the 4 row-subgroups (lanes l, l+8, l+16, l+24 share columns)
    #pragma unroll
    for (int j = 0; j < 8; ++j) {
        acc[j] += __shfl_xor(acc[j], 8, 32);
        acc[j] += __shfl_xor(acc[j], 16, 32);
    }

    if (lane < 8) {
        const int d = dst[e];
        float* out = agg + (size_t)d * OUT + lane * 8;
        #pragma unroll
        for (int j = 0; j < 8; ++j)
            atomicAdd(out + j, acc[j]);   // global_atomic_add_f32, coalesced
    }
}

// ---------------------------------------------------------------------------
// h_next = relu(agg + bias)
// ---------------------------------------------------------------------------
__global__ void bias_relu(const float* __restrict__ agg,
                          const float* __restrict__ bias,
                          float* __restrict__ out, int n) {
    int idx = blockIdx.x * blockDim.x + threadIdx.x;
    if (idx < n) out[idx] = fmaxf(agg[idx] + bias[idx & 63], 0.0f);
}

// ---------------------------------------------------------------------------
extern "C" void kernel_launch(void* const* d_in, const int* in_sizes, int n_in,
                              void* d_out, int out_size, void* d_ws, size_t ws_size,
                              hipStream_t stream) {
    const float* nf   = (const float*)d_in[0];
    const float* ef   = (const float*)d_in[1];
    const int*   src  = (const int*)  d_in[2];
    const int*   dst  = (const int*)  d_in[3];
    const float* Wp   = (const float*)d_in[4];
    const float* bp   = (const float*)d_in[5];
    const float* We1  = (const float*)d_in[6];
    const float* be1  = (const float*)d_in[7];
    const float* We2  = (const float*)d_in[8];
    const float* be2  = (const float*)d_in[9];
    const float* bias = (const float*)d_in[10];

    const int V = in_sizes[0] / NODE_IN;
    const int E = in_sizes[2];

    char* ws = (char*)d_ws;
    size_t off = 0;
    _Float16* ew = (_Float16*)(ws + off); off += (size_t)E * OUTSQ * sizeof(_Float16);
    _Float16* h1 = (_Float16*)(ws + off); off += (size_t)E * EHID  * sizeof(_Float16);
    _Float16* Bt = (_Float16*)(ws + off); off += (size_t)EHID * OUTSQ * sizeof(_Float16);
    float*    h  = (float*)   (ws + off); off += (size_t)V * OUT * sizeof(float);
    float*    agg= (float*)   (ws + off);

    const int nVO = V * OUT;

    // 1) project node features
    proj_nodes<<<(nVO + 255) / 256, 256, 0, stream>>>(nf, Wp, bp, h, V);
    // 2) edge MLP layer 1 (f16 A operand)
    edge_layer1<<<((size_t)E * EHID + 255) / 256, 256, 0, stream>>>(ef, We1, be1, h1, E);
    // 3) transpose We2 -> Bt (f16 B operand)
    transpose_We2<<<(EHID * OUTSQ) / 256, 256, 0, stream>>>(We2, Bt);
    // 4) big WMMA GEMM: ew = h1 @ We2 + be2  (f16 output, 800 MB)
    {
        dim3 grid((E + 15) / 16, OUTSQ / 512);
        ew_gemm<<<grid, 256, 0, stream>>>(h1, Bt, be2, ew, E);
    }
    // 5) message-passing steps
    const int msg_blocks = (E * 32 + 255) / 256;   // one wave per edge
    for (int step = 0; step < NUM_STEPS; ++step) {
        zero_f32<<<(nVO + 255) / 256, 256, 0, stream>>>(agg, nVO);
        msg_scatter<<<msg_blocks, 256, 0, stream>>>(h, ew, src, dst, agg, E);
        float* out = (step == NUM_STEPS - 1) ? (float*)d_out : h;
        bias_relu<<<(nVO + 255) / 256, 256, 0, stream>>>(agg, bias, out, nVO);
    }
}
